// Net_17188459119113
// MI455X (gfx1250) — 8-bit matrix operand order repaired and checked by reading, compiled, not run on hardware
//
#include <hip/hip_runtime.h>
#include <hip/hip_bf16.h>

typedef __attribute__((ext_vector_type(8))) int v8i;


__device__ __forceinline__ float qclamp7(float v) {
    return fminf(fmaxf(v, -7.f), 7.f);
}

__global__ void k_init(unsigned int* ws_u) {
    if (threadIdx.x == 0) { ws_u[0] = 0u; ws_u[1] = 0u; }
}

__global__ void k_max_abs(const float* __restrict__ x, int n, unsigned int* ws_u) {
    __shared__ float red[256];
    float m = 0.f;
    for (int i = blockIdx.x * blockDim.x + threadIdx.x; i < n; i += gridDim.x * blockDim.x)
        m = fmaxf(m, fabsf(x[i]));
    red[threadIdx.x] = m;
    __syncthreads();
    for (int s = 128; s > 0; s >>= 1) {
        if (threadIdx.x < s) red[threadIdx.x] = fmaxf(red[threadIdx.x], red[threadIdx.x + s]);
        __syncthreads();
    }
    if (threadIdx.x == 0) atomicMax(&ws_u[0], __float_as_uint(red[0]));
}

__global__ void k_quant_weights(const float* __restrict__ cw, const float* __restrict__ fw,
                                const unsigned int* __restrict__ ws_u, float* __restrict__ wsf,
                                int* __restrict__ qcw, signed char* __restrict__ qfcw) {
    __shared__ float red[256];
    __shared__ float s_cw_sh, s_fw_sh;
    const int t = threadIdx.x;
    red[t] = (t < 144) ? fabsf(cw[t]) : 0.f;
    __syncthreads();
    for (int s = 128; s > 0; s >>= 1) {
        if (t < s) red[t] = fmaxf(red[t], red[t + s]);
        __syncthreads();
    }
    if (t == 0) s_cw_sh = red[0] / 7.f + 1e-8f;
    __syncthreads();
    float m = 0.f;
    for (int i = t; i < 10 * 576; i += 256) m = fmaxf(m, fabsf(fw[i]));
    red[t] = m;
    __syncthreads();
    for (int s = 128; s > 0; s >>= 1) {
        if (t < s) red[t] = fmaxf(red[t], red[t + s]);
        __syncthreads();
    }
    if (t == 0) s_fw_sh = red[0] / 7.f + 1e-8f;
    __syncthreads();
    const float s_cw = s_cw_sh, s_fw = s_fw_sh;
    if (t == 0) {
        wsf[0] = __uint_as_float(ws_u[0]) / 7.f + 1e-8f;
        wsf[1] = s_cw;
        wsf[2] = s_fw;
    }
    if (t < 144) qcw[t] = (int)rintf(qclamp7(cw[t] / s_cw));
    for (int i = t; i < 16 * 576; i += 256) {
        const int r = i / 576, c = i % 576;
        signed char q = 0;
        if (r < 10) q = (signed char)(int)rintf(qclamp7(fw[r * 576 + c] / s_fw));
        qfcw[i] = q;
    }
}

__global__ void k_scale_f(const unsigned int* __restrict__ ws_u, float* __restrict__ wsf) {
    wsf[3] = __uint_as_float(ws_u[1]) / 7.f + 1e-8f;
}

__global__ void k_conv(const float* __restrict__ x, const float* __restrict__ cb_g,
                       const float* __restrict__ wsf, const int* __restrict__ qcw_g,
                       unsigned int* __restrict__ ws_u, signed char* __restrict__ qflat,
                       int pass) {
    __shared__ int   qx[784];
    __shared__ int   qcw[144];
    __shared__ float cb[16];
    __shared__ float pm[576];
    const int t = threadIdx.x;
    const int img = blockIdx.x;
    const float s_x = wsf[0];
    const float sxw = s_x * wsf[1];
    const float* xi = x + (size_t)img * 784;
    for (int i = t; i < 784; i += 576) qx[i] = (int)rintf(qclamp7(xi[i] / s_x));
    if (t < 144) qcw[t] = qcw_g[t];
    if (t < 16)  cb[t] = cb_g[t];
    __syncthreads();

    const int c = t / 36, p = t % 36, py = p / 6, px = p % 6;
    const int* w = &qcw[c * 9];
    float m = 0.f;
    for (int dy = 0; dy < 4; ++dy)
        for (int dx = 0; dx < 4; ++dx) {
            const int oy = py * 4 + dy, ox = px * 4 + dx;
            int sum = 0;
#pragma unroll
            for (int ky = 0; ky < 3; ++ky)
#pragma unroll
                for (int kx = 0; kx < 3; ++kx)
                    sum += qx[(oy + ky) * 28 + (ox + kx)] * w[ky * 3 + kx];
            m = fmaxf(m, sxw * (float)sum + cb[c]);
        }

    if (pass == 0) {
        pm[t] = m;
        __syncthreads();
        for (int s = 512; s > 0; s >>= 1) {
            if (t < s && t + s < 576) pm[t] = fmaxf(pm[t], pm[t + s]);
            __syncthreads();
        }
        if (t == 0) atomicMax(&ws_u[1], __float_as_uint(pm[0]));
    } else {
        const float s_f = wsf[3];
        qflat[(size_t)img * 576 + t] = (signed char)(int)rintf(qclamp7(m / s_f));
    }
}

__global__ void k_fc(const signed char* __restrict__ qflat, const signed char* __restrict__ qfcw,
                     const float* __restrict__ fcb, const float* __restrict__ wsf,
                     float* __restrict__ out) {
    const int lane = threadIdx.x & 31;
    const int wave = threadIdx.x >> 5;
    const int tile = blockIdx.x * (blockDim.x >> 5) + wave;
    const int row0 = tile * 16;
    const int half = lane >> 4;
    const int l15 = lane & 15;

    const unsigned int* arow = (const unsigned int*)(qflat + (size_t)(row0 + l15) * 576);
    const unsigned int* brow = (const unsigned int*)(qfcw + (size_t)l15 * 576);

    v8i acc = {};
    for (int kt = 0; kt < 9; ++kt) {
        const int kb = kt * 64;
        v8i a, b;
#pragma unroll
        for (int v = 0; v < 8; ++v) {
            const int ka = (v >> 2) * 32 + ((v >> 1) & 1) * 16 + half * 8 + (v & 1) * 4;
            a[v] = (int)arow[(kb + ka) >> 2];
            const int kbb = (v >> 2) * 32 + ((v >> 1) & 1) * 16 + half * 8 + (v & 1) * 4;
            b[v] = (int)brow[(kb + kbb) >> 2];
        }
        acc = __builtin_amdgcn_wmma_i32_16x16x64_iu8(true, a, true, b, acc, false, false);
    }

    const float scale = wsf[3] * wsf[2];
    const int n = l15;
    const float bias = (n < 10) ? fcb[n] : 0.f;
#pragma unroll
    for (int r = 0; r < 8; ++r) {
        const int mrow = r + 8 * half;
        if (n < 10)
            out[(size_t)(row0 + mrow) * 10 + n] = scale * (float)acc[r] + bias;
    }
}

extern "C" void kernel_launch(void* const* d_in, const int* in_sizes, int n_in,
                              void* d_out, int out_size, void* d_ws, size_t ws_size,
                              hipStream_t stream) {
    const float* x   = (const float*)d_in[0];
    const float* cw  = (const float*)d_in[1];
    const float* cb  = (const float*)d_in[2];
    const float* fw  = (const float*)d_in[3];
    const float* fcb = (const float*)d_in[4];
    float* out = (float*)d_out;

    char* ws = (char*)d_ws;
    unsigned int* ws_u = (unsigned int*)ws;
    float*        wsf  = (float*)(ws + 8);
    int*          qcw  = (int*)(ws + 64);
    signed char*  qfcw = (signed char*)(ws + 1024);
    signed char*  qflat= (signed char*)(ws + 16384);

    const int nx = in_sizes[0];
    const int B  = nx / 784;

    hipLaunchKernelGGL(k_init,          dim3(1),       dim3(32),  0, stream, ws_u);
    hipLaunchKernelGGL(k_max_abs,       dim3(512),     dim3(256), 0, stream, x, nx, ws_u);
    hipLaunchKernelGGL(k_quant_weights, dim3(1),       dim3(256), 0, stream, cw, fw, ws_u, wsf, qcw, qfcw);
    hipLaunchKernelGGL(k_conv,          dim3(B),       dim3(576), 0, stream, x, cb, wsf, qcw, ws_u, qflat, 0);
    hipLaunchKernelGGL(k_scale_f,       dim3(1),       dim3(1),   0, stream, ws_u, wsf);
    hipLaunchKernelGGL(k_conv,          dim3(B),       dim3(576), 0, stream, x, cb, wsf, qcw, ws_u, qflat, 1);
    hipLaunchKernelGGL(k_fc,            dim3(B / 128), dim3(256), 0, stream, qflat, qfcw, fcb, wsf, out);
}
